// LatentGraphNetwork_37701222924777
// MI455X (gfx1250) — compile-verified, matrix operand order checked by reading, not run on hardware
//
#include <hip/hip_runtime.h>
#include <cstdint>

// Problem dims
#define NTOK   (32 * 4096)
#define D      256
#define MTILE  128   // tokens per workgroup

typedef __bf16        v16bf __attribute__((ext_vector_type(16)));
typedef float         v8f   __attribute__((ext_vector_type(8)));
typedef float         f4    __attribute__((ext_vector_type(4)));
typedef unsigned int  u4    __attribute__((ext_vector_type(4)));
typedef unsigned int  u2    __attribute__((ext_vector_type(2)));

// ---------- helpers ----------
static __device__ __forceinline__ unsigned short f2bf(float f) {
    unsigned u = __float_as_uint(f);
    unsigned r = u + 0x7FFFu + ((u >> 16) & 1u);   // round-to-nearest-even
    return (unsigned short)(r >> 16);
}
static __device__ __forceinline__ unsigned pack2(float lo, float hi) {
    return (unsigned)f2bf(lo) | ((unsigned)f2bf(hi) << 16);
}
static __device__ __forceinline__ float gelu_erf(float x) {
    return 0.5f * x * (1.0f + erff(x * 0.70710678118654752f));
}

// CDNA5 async global->LDS copy of a 128KB weight panel (256x256 bf16).
// Each thread issues 32 x b128 async copies; tracked on ASYNCcnt.
static __device__ __forceinline__ void async_copy_w(unsigned short* sdst,
                                                    const unsigned short* gsrc,
                                                    int tid) {
    unsigned ldsbase = (unsigned)(uintptr_t)(void*)sdst;  // low 32 bits = LDS byte addr
    const char* g = (const char*)gsrc;
    #pragma unroll
    for (int i = 0; i < 32; ++i) {
        unsigned off  = (unsigned)((i * 256 + tid) * 16);
        unsigned ldsa = ldsbase + off;
        const void* ga = g + off;
        asm volatile("global_load_async_to_lds_b128 %0, %1, off"
                     :: "v"(ldsa), "v"(ga) : "memory");
    }
}
static __device__ __forceinline__ void wait_async0() {
    asm volatile("s_wait_asynccnt 0" ::: "memory");
}

// Load one 16x32 bf16 fragment (A-layout, ISA 7.12.2) from LDS, row-major stride D.
static __device__ __forceinline__ v16bf load_frag(const unsigned short* base,
                                                  int row, int kbase, int lane) {
    const unsigned short* p = base + row * D + kbase + ((lane >> 4) << 3);
    union { u4 q[2]; v16bf v; } u;
    u.q[0] = *(const u4*)p;          // K = kbase+khalf .. +7   -> VGPR 0..3
    u.q[1] = *(const u4*)(p + 16);   // K = +16 .. +23          -> VGPR 4..7
    return u.v;
}

// One K=256 GEMM producing this wave's 32x128 output tile.
// Rows [rbase, rbase+32), cols [cbase, cbase+128). A from sA, weight W[out][in] from sW.
// Per K-step: 4 A-loads + 16 B-loads feed 16 WMMAs (B reused 2x, A reused 8x).
static __device__ __forceinline__ void gemm_tile(v8f acc[2][8],
                                                 const unsigned short* sA,
                                                 const unsigned short* sW,
                                                 int rbase, int cbase, int lane) {
    const int ar = rbase + (lane & 15);
    const int wr = cbase + (lane & 15);
    for (int kt = 0; kt < 8; ++kt) {
        v16bf a0 = load_frag(sA, ar,      kt * 32, lane);
        v16bf a1 = load_frag(sA, ar + 16, kt * 32, lane);
        #pragma unroll
        for (int nt = 0; nt < 8; ++nt) {
            v16bf b = load_frag(sW, wr + nt * 16, kt * 32, lane);
            acc[0][nt] = __builtin_amdgcn_wmma_f32_16x16x32_bf16(
                false, a0, false, b, (short)0, acc[0][nt], false, false);
            acc[1][nt] = __builtin_amdgcn_wmma_f32_16x16x32_bf16(
                false, a1, false, b, (short)0, acc[1][nt], false, false);
        }
    }
}

static __device__ __forceinline__ void zero_acc(v8f acc[2][8]) {
    #pragma unroll
    for (int rt = 0; rt < 2; ++rt)
        #pragma unroll
        for (int nt = 0; nt < 8; ++nt)
            acc[rt][nt] = (v8f){0,0,0,0,0,0,0,0};
}

// ---------- kernel A: weight prep ----------
// bf16-convert msg_W / upd_W[:, :256] / out_W, and compute the fused weight
// Wf[o][h] = sum_g upd_W[o][256+g] * graph[h][g]  (folds the 'agg' GEMM away).
__global__ void prep_weights(const float* __restrict__ graph,
                             const float* __restrict__ msg_W,
                             const float* __restrict__ upd_W,
                             const float* __restrict__ out_W,
                             unsigned short* __restrict__ wWmsg,
                             unsigned short* __restrict__ wW1,
                             unsigned short* __restrict__ wWf,
                             unsigned short* __restrict__ wWout) {
    const int o = blockIdx.x;   // output feature
    const int t = threadIdx.x;  // 0..255
    wWmsg[o * 256 + t] = f2bf(msg_W[o * 256 + t]);
    wW1  [o * 256 + t] = f2bf(upd_W[o * 512 + t]);
    wWout[o * 256 + t] = f2bf(out_W[o * 256 + t]);

    __shared__ float w2[256];
    __shared__ __align__(16) float sG[256 * 65];  // 64-col graph chunk, padded
    w2[t] = upd_W[o * 512 + 256 + t];
    float s = 0.0f;
    for (int g0 = 0; g0 < 256; g0 += 64) {
        __syncthreads();
        #pragma unroll
        for (int i = 0; i < 64; ++i) {          // coalesced load of graph[:, g0:g0+64]
            int e = i * 256 + t;
            int h = e >> 6, c = e & 63;
            sG[h * 65 + c] = graph[h * 256 + g0 + c];
        }
        __syncthreads();
        #pragma unroll
        for (int c = 0; c < 64; ++c)            // thread t == row h
            s = fmaf(w2[g0 + c], sG[t * 65 + c], s);
    }
    wWf[o * 256 + t] = f2bf(s);
}

// ---------- kernel B: fully fused network ----------
__global__ __launch_bounds__(256, 1)
void lgn_fused(const float* __restrict__ x,
               const float* __restrict__ msg_b,
               const float* __restrict__ upd_b,
               const float* __restrict__ out_b,
               const float* __restrict__ ln_g,
               const float* __restrict__ ln_b,
               const unsigned short* __restrict__ wWmsg,
               const unsigned short* __restrict__ wW1,
               const unsigned short* __restrict__ wWf,
               const unsigned short* __restrict__ wWout,
               float* __restrict__ out) {
    __shared__ __align__(16) unsigned short sX[MTILE * D];   // 64KB: x tile, later h tile
    __shared__ __align__(16) unsigned short sMsg[MTILE * D]; // 64KB: msg tile
    __shared__ __align__(16) unsigned short sW[D * D];       // 128KB: staged weights / f32 out

    const int tid   = threadIdx.x;
    const int wv    = tid >> 5;
    const int lane  = tid & 31;
    const int rbase = (wv & 3) * 32;   // row quarter (32 rows)
    const int cbase = (wv >> 2) * 128; // column half (128 cols)
    const int col0  = lane & 15;
    const int rhalf = (lane >> 4) << 3;
    const size_t tok0 = (size_t)blockIdx.x * MTILE;

    // ---- stage 0: async-stage msg_W while loading+converting the x tile ----
    async_copy_w(sW, wWmsg, tid);
    {
        const f4* xg = (const f4*)(x + tok0 * D);
        #pragma unroll
        for (int i = 0; i < 32; ++i) {
            int c = i * 256 + tid;             // float4 chunk index
            f4 v = xg[c];
            u2 p; p.x = pack2(v.x, v.y); p.y = pack2(v.z, v.w);
            *(u2*)(sX + c * 4) = p;
        }
    }
    wait_async0();
    __syncthreads();

    // ---- stage 1: msg = x @ msg_W^T + msg_b  (bf16 -> sMsg) ----
    {
        v8f acc[2][8];
        zero_acc(acc);
        gemm_tile(acc, sX, sW, rbase, cbase, lane);
        __syncthreads();                  // all waves done reading Wmsg
        async_copy_w(sW, wW1, tid);       // W1 load overlaps epilogue below
        #pragma unroll
        for (int rt = 0; rt < 2; ++rt)
            #pragma unroll
            for (int nt = 0; nt < 8; ++nt) {
                int c = cbase + nt * 16 + col0;
                float bv = msg_b[c];
                #pragma unroll
                for (int j = 0; j < 8; ++j)
                    sMsg[(rbase + rt * 16 + rhalf + j) * D + c] = f2bf(acc[rt][nt][j] + bv);
            }
        wait_async0();
        __syncthreads();

        // ---- stage 2: h = gelu(x @ W1^T + msg @ Wf^T + upd_b) (overwrites sX) ----
        zero_acc(acc);
        gemm_tile(acc, sX, sW, rbase, cbase, lane);   // x part
        __syncthreads();                  // everyone done with sX (as input) and W1
        async_copy_w(sW, wWf, tid);
        wait_async0();
        __syncthreads();
        gemm_tile(acc, sMsg, sW, rbase, cbase, lane); // folded-graph msg part
        __syncthreads();                  // all waves done reading Wf
        async_copy_w(sW, wWout, tid);     // Wout load overlaps GELU epilogue below
        #pragma unroll
        for (int rt = 0; rt < 2; ++rt)
            #pragma unroll
            for (int nt = 0; nt < 8; ++nt) {
                int c = cbase + nt * 16 + col0;
                float bv = upd_b[c];
                #pragma unroll
                for (int j = 0; j < 8; ++j)
                    sX[(rbase + rt * 16 + rhalf + j) * D + c] =
                        f2bf(gelu_erf(acc[rt][nt][j] + bv));
            }
        wait_async0();
        __syncthreads();

        // ---- stage 3: out = h @ out_W^T + out_b  (f32 tile into sW) ----
        zero_acc(acc);
        gemm_tile(acc, sX, sW, rbase, cbase, lane);
        __syncthreads();                  // all waves done reading Wout
        float* sOutF = (float*)sW;        // reuse 128KB = 128x256 f32
        #pragma unroll
        for (int rt = 0; rt < 2; ++rt)
            #pragma unroll
            for (int nt = 0; nt < 8; ++nt) {
                int c = cbase + nt * 16 + col0;
                float bv = out_b[c];
                #pragma unroll
                for (int j = 0; j < 8; ++j)
                    sOutF[(rbase + rt * 16 + rhalf + j) * D + c] = acc[rt][nt][j] + bv;
            }
    }
    __syncthreads();

    // ---- LayerNorm(256) + store: 2 threads per row, wave32 shfl reduction ----
    {
        const float* sOutF = (const float*)sW;
        int r  = tid >> 1;
        int hh = tid & 1;
        const float* rp = sOutF + r * D + hh * 128;
        float s = 0.0f, s2 = 0.0f;
        #pragma unroll
        for (int i = 0; i < 128; i += 4) {
            f4 v = *(const f4*)(rp + i);
            s  += v.x + v.y + v.z + v.w;
            s2 += v.x * v.x + v.y * v.y + v.z * v.z + v.w * v.w;
        }
        s  += __shfl_xor(s, 1, 32);
        s2 += __shfl_xor(s2, 1, 32);
        float mean = s * (1.0f / 256.0f);
        float var  = s2 * (1.0f / 256.0f) - mean * mean;
        float inv  = rsqrtf(var + 1e-5f);
        float* og = out + (tok0 + (size_t)r) * D + hh * 128;
        const float* gg = ln_g + hh * 128;
        const float* bb = ln_b + hh * 128;
        #pragma unroll
        for (int i = 0; i < 128; i += 4) {
            f4 v = *(const f4*)(rp + i);
            f4 o;
            o.x = (v.x - mean) * inv * gg[i + 0] + bb[i + 0];
            o.y = (v.y - mean) * inv * gg[i + 1] + bb[i + 1];
            o.z = (v.z - mean) * inv * gg[i + 2] + bb[i + 2];
            o.w = (v.w - mean) * inv * gg[i + 3] + bb[i + 3];
            *(f4*)(og + i) = o;
        }
    }
}

// ---------- launch ----------
extern "C" void kernel_launch(void* const* d_in, const int* in_sizes, int n_in,
                              void* d_out, int out_size, void* d_ws, size_t ws_size,
                              hipStream_t stream) {
    (void)in_sizes; (void)n_in; (void)out_size; (void)ws_size;
    const float* x     = (const float*)d_in[0];
    const float* graph = (const float*)d_in[1];
    const float* msg_W = (const float*)d_in[2];
    const float* msg_b = (const float*)d_in[3];
    const float* upd_W = (const float*)d_in[4];
    const float* upd_b = (const float*)d_in[5];
    const float* out_W = (const float*)d_in[6];
    const float* out_b = (const float*)d_in[7];
    const float* ln_g  = (const float*)d_in[8];
    const float* ln_b  = (const float*)d_in[9];
    float* out = (float*)d_out;

    unsigned short* wWmsg = (unsigned short*)d_ws;      // 4 x 128KB bf16 panels
    unsigned short* wW1   = wWmsg + 65536;
    unsigned short* wWf   = wW1   + 65536;
    unsigned short* wWout = wWf   + 65536;

    prep_weights<<<256, 256, 0, stream>>>(graph, msg_W, upd_W, out_W,
                                          wWmsg, wW1, wWf, wWout);
    lgn_fused<<<NTOK / MTILE, 256, 0, stream>>>(x, msg_b, upd_b, out_b, ln_g, ln_b,
                                                wWmsg, wW1, wWf, wWout, out);
}
